// BiLSTMEncoder_31533649887749
// MI455X (gfx1250) — matrix operand order repaired and checked by reading, not compiled, not run on hardware
//
#include <hip/hip_runtime.h>

typedef __attribute__((ext_vector_type(16))) _Float16 v16h;
typedef __attribute__((ext_vector_type(8)))  float    v8f;

#define L_SEQ  512
#define BATCH  128
#define EMBED  64
#define HIDDEN 128
#define NGATES 512   // 4*HIDDEN
#define PROJ   64

// ---- async global->LDS path (CDNA5 GLOBAL_LOAD_ASYNC_TO_LDS_B128) ----------
#if defined(__has_builtin)
#  if __has_builtin(__builtin_amdgcn_global_load_async_to_lds_b128)
#    define ASYNC_LDS 1
#  endif
#endif
#ifndef ASYNC_LDS
#  define ASYNC_LDS 0
#endif

// builtin signature is (v4i addrspace(1)*, v4i addrspace(3)*, imm int, imm int)
typedef int b128v __attribute__((vector_size(4 * sizeof(int))));
typedef b128v __attribute__((address_space(1))) gb128_t;
typedef b128v __attribute__((address_space(3))) lb128_t;

__device__ __forceinline__ void async_copy_b128(const void* g, void* l) {
#if ASYNC_LDS
  // generic LDS pointer: low 32 bits are the LDS byte offset
  __builtin_amdgcn_global_load_async_to_lds_b128(
      (gb128_t*)(uintptr_t)g, (lb128_t*)(unsigned)(uintptr_t)l, 0, 0);
#else
  (void)g; (void)l;
#endif
}
__device__ __forceinline__ void wait_async0() {
#if ASYNC_LDS
#  if __has_builtin(__builtin_amdgcn_s_wait_asynccnt)
  __builtin_amdgcn_s_wait_asynccnt(0);
#  else
  asm volatile("s_wait_asynccnt 0" ::: "memory");
#  endif
#endif
}

struct DirParams {
  const float* Wih;  // (4H, IN)
  const float* Whh;  // (4H, PROJ)
  const float* bih;  // (4H)
  const float* bhh;  // (4H)
  const float* Whr;  // (PROJ, HIDDEN)
};

union FragH { v16h v; uint4 u[2]; };

// fast activations: v_exp_f32 + v_rcp_f32 (no precise-division expansion)
__device__ __forceinline__ float sigmoid_(float x) {
  return __builtin_amdgcn_rcpf(1.0f + __expf(-x));
}
__device__ __forceinline__ float tanh_(float x) {
  // tanh(x) = 2*sigmoid(2x) - 1
  return __builtin_amdgcn_rcpf(1.0f + __expf(-2.0f * x)) * 2.0f - 1.0f;
}

// A-fragment (16x32 f16) from LDS row-major matrix, leading dim `ld` halves.
// lane<16: halves K = k0+0..7 , k0+16..23 ; lane>=16: +8 offset on both runs.
__device__ __forceinline__ v16h load_A(const _Float16* base, int ld, int m0, int k0, int lane) {
  const _Float16* row = base + (m0 + (lane & 15)) * ld + k0 + ((lane >= 16) ? 8 : 0);
  FragH f;
  f.u[0] = *(const uint4*)(row);
  f.u[1] = *(const uint4*)(row + 16);
  return f.v;
}

// B-fragment (32x16 f16): B(k,n) = W[n][k], W row-major [n][ld].
// lane holds column n0+(lane&15); K = k0+0..7, k0+16..23 ; lane>=16: +8 offset on both runs.
__device__ __forceinline__ v16h load_B(const _Float16* base, int ld, int n0, int k0, int lane) {
  const _Float16* row = base + (n0 + (lane & 15)) * ld + k0 + ((lane >= 16) ? 8 : 0);
  FragH f;
  f.u[0] = *(const uint4*)(row);
  f.u[1] = *(const uint4*)(row + 16);
  return f.v;
}

// ---------------------------------------------------------------------------
// Embedding gather + f32 -> f16 convert.  One thread = 8 halves of one token.
// ---------------------------------------------------------------------------
__global__ __launch_bounds__(256)
void embed_f16_kernel(const int* __restrict__ tok,
                      const float* __restrict__ emb,
                      _Float16* __restrict__ X0) {
  int gid = blockIdx.x * 256 + threadIdx.x;     // L*B*8 threads
  int pos = gid >> 3;                           // 0 .. L*B-1
  int seg = (gid & 7) * 8;                      // 8-half segment within EMBED
  int tk  = tok[pos];
  const float4* s = (const float4*)(emb + (size_t)tk * EMBED + seg);
  float4 v0 = s[0], v1 = s[1];
  union { _Float16 h[8]; uint4 u; } o;
  o.h[0] = (_Float16)v0.x; o.h[1] = (_Float16)v0.y;
  o.h[2] = (_Float16)v0.z; o.h[3] = (_Float16)v0.w;
  o.h[4] = (_Float16)v1.x; o.h[5] = (_Float16)v1.y;
  o.h[6] = (_Float16)v1.z; o.h[7] = (_Float16)v1.w;
  *(uint4*)(X0 + (size_t)pos * EMBED + seg) = o.u;
}

// ---------------------------------------------------------------------------
// Persistent bidirectional LSTM scan for one layer.
// grid = 2 (blockIdx.x = direction), block = 512 threads = 16 wave32.
// Weights f16-resident in LDS; cell state c in VGPRs; x_t double-buffered
// with async global->LDS prefetch overlapping the WMMA work.
// Gate chunks processed in two groups of two (i,f then g,o); accumulators
// start at zero (rematerializable, inline-0 C operand) and the bias is
// added post-GEMM so no large loop-invariant vectors get hoisted + spilled.
// ---------------------------------------------------------------------------
template<int IN>
__global__ __launch_bounds__(512)
void lstm_scan_kernel(DirParams pf, DirParams pb,
                      const _Float16* __restrict__ xseq,   // (L, B, IN) f16
                      _Float16* __restrict__ outH,         // (L, B, 2*PROJ) f16 or null
                      float* __restrict__ outF) {          // (L, B, 2*PROJ) f32 or null
  extern __shared__ char smem[];
  const int dir = blockIdx.x;
  const DirParams P = dir ? pb : pf;

  _Float16* sWih = (_Float16*)smem;                   // 512*IN
  _Float16* sWhh = sWih + NGATES * IN;                // 512*64
  _Float16* sWhr = sWhh + NGATES * PROJ;              // 64*128
  _Float16* sX   = sWhr + PROJ * HIDDEN;              // 2 * 128*IN (ping-pong)
  _Float16* sH   = sX + 2 * BATCH * IN;               // 128*64   (projected h)
  _Float16* sA   = sH + BATCH * PROJ;                 // 128*128  (o*tanh(c))

  const int tid  = threadIdx.x;
  const int lane = tid & 31;
  const int wave = tid >> 5;

  // ---- stage weights once (f32 -> f16), zero h ----
  for (int i = tid; i < NGATES * IN / 4; i += 512) {
    float4 v = ((const float4*)P.Wih)[i];
    union { _Float16 h[4]; uint2 u; } c;
    c.h[0] = (_Float16)v.x; c.h[1] = (_Float16)v.y;
    c.h[2] = (_Float16)v.z; c.h[3] = (_Float16)v.w;
    ((uint2*)sWih)[i] = c.u;
  }
  for (int i = tid; i < NGATES * PROJ / 4; i += 512) {
    float4 v = ((const float4*)P.Whh)[i];
    union { _Float16 h[4]; uint2 u; } c;
    c.h[0] = (_Float16)v.x; c.h[1] = (_Float16)v.y;
    c.h[2] = (_Float16)v.z; c.h[3] = (_Float16)v.w;
    ((uint2*)sWhh)[i] = c.u;
  }
  for (int i = tid; i < PROJ * HIDDEN / 4; i += 512) {
    float4 v = ((const float4*)P.Whr)[i];
    union { _Float16 h[4]; uint2 u; } c;
    c.h[0] = (_Float16)v.x; c.h[1] = (_Float16)v.y;
    c.h[2] = (_Float16)v.z; c.h[3] = (_Float16)v.w;
    ((uint2*)sWhr)[i] = c.u;
  }
  {
    uint4 z; z.x = 0; z.y = 0; z.z = 0; z.w = 0;
    for (int i = tid; i < BATCH * PROJ / 8; i += 512) ((uint4*)sH)[i] = z;
  }

  // ---- gate bias in registers: bias[ch*128 + ct*16 + (lane&15)] ----
  float biasr[4][4];
#pragma unroll
  for (int p = 0; p < 4; ++p) {
    const int ct = (wave * 4 + p) & 7;
#pragma unroll
    for (int ch = 0; ch < 4; ++ch) {
      const int g = ch * HIDDEN + ct * 16 + (lane & 15);
      biasr[p][ch] = P.bih[g] + P.bhh[g];
    }
  }

  // cell state: wave owns 4 (mt,ct) tile-pairs; each holds a v8f c fragment
  v8f cst[4] = { {}, {}, {}, {} };

#if ASYNC_LDS
  {  // prefetch x for the first step into buffer 0
    const int t0 = dir ? (L_SEQ - 1) : 0;
    const char* src = (const char*)(xseq + (size_t)t0 * BATCH * IN);
    for (int i = tid; i < BATCH * IN / 8; i += 512)
      async_copy_b128(src + i * 16, (char*)sX + i * 16);
  }
#endif
  __syncthreads();

  for (int step = 0; step < L_SEQ; ++step) {
    const int t = dir ? (L_SEQ - 1 - step) : step;

#if ASYNC_LDS
    wait_async0();           // x_t fill (issued last step) complete
    __syncthreads();         // publish to all waves
    if (step + 1 < L_SEQ) {  // overlap next fill with this step's WMMA work
      const int tn = dir ? (L_SEQ - 2 - step) : (step + 1);
      const char* src = (const char*)(xseq + (size_t)tn * BATCH * IN);
      char* dst = (char*)(sX + ((step + 1) & 1) * BATCH * IN);
      for (int i = tid; i < BATCH * IN / 8; i += 512)
        async_copy_b128(src + i * 16, dst + i * 16);
    }
    const _Float16* sXc = sX + (step & 1) * BATCH * IN;
#else
    {
      const uint4* src = (const uint4*)(xseq + (size_t)t * BATCH * IN);
      uint4* dst = (uint4*)sX;
      for (int i = tid; i < BATCH * IN / 8; i += 512) dst[i] = src[i];
    }
    __syncthreads();
    const _Float16* sXc = sX;
#endif

    // ---- gate GEMM: gates(128 x 512) = x_t @ Wih^T + h @ Whh^T + bias ----
    // wave handles 4 (mt,ct) pairs; each pair = 4 gate chunks (i,f,g,o) at
    // the same (m,col) so the cell update is lane-local.  Chunks are done
    // in two groups of two to limit live accumulators; bias added post-GEMM.
#pragma unroll
    for (int p = 0; p < 4; ++p) {
      const int pair = wave * 4 + p;
      const int mt = pair >> 3, ct = pair & 7;
      const int m0 = mt * 16;
      const int col = ct * 16 + (lane & 15);          // cell column 0..127
      const int rowOff = (lane >= 16) ? 8 : 0;

      float ig[8], fc[8];                             // carries between groups

      // ---- group 0: chunks 0 (i) and 1 (f) ----
      {
        v8f acc0 = {}, acc1 = {};
#pragma unroll
        for (int k0 = 0; k0 < IN; k0 += 32) {
          v16h a = load_A(sXc, IN, m0, k0, lane);
          v16h b0 = load_B(sWih, IN, 0 * HIDDEN + ct * 16, k0, lane);
          acc0 = __builtin_amdgcn_wmma_f32_16x16x32_f16(
              false, a, false, b0, (short)0, acc0, false, false);
          v16h b1 = load_B(sWih, IN, 1 * HIDDEN + ct * 16, k0, lane);
          acc1 = __builtin_amdgcn_wmma_f32_16x16x32_f16(
              false, a, false, b1, (short)0, acc1, false, false);
        }
#pragma unroll
        for (int k0 = 0; k0 < PROJ; k0 += 32) {
          v16h a = load_A(sH, PROJ, m0, k0, lane);
          v16h b0 = load_B(sWhh, PROJ, 0 * HIDDEN + ct * 16, k0, lane);
          acc0 = __builtin_amdgcn_wmma_f32_16x16x32_f16(
              false, a, false, b0, (short)0, acc0, false, false);
          v16h b1 = load_B(sWhh, PROJ, 1 * HIDDEN + ct * 16, k0, lane);
          acc1 = __builtin_amdgcn_wmma_f32_16x16x32_f16(
              false, a, false, b1, (short)0, acc1, false, false);
        }
#pragma unroll
        for (int e = 0; e < 8; ++e) {
          ig[e] = sigmoid_(acc0[e] + biasr[p][0]);               // input gate
          fc[e] = sigmoid_(acc1[e] + biasr[p][1]) * cst[p][e];   // f * c
        }
      }

      // ---- group 1: chunks 2 (g) and 3 (o) + cell update ----
      {
        v8f acc0 = {}, acc1 = {};
#pragma unroll
        for (int k0 = 0; k0 < IN; k0 += 32) {
          v16h a = load_A(sXc, IN, m0, k0, lane);
          v16h b0 = load_B(sWih, IN, 2 * HIDDEN + ct * 16, k0, lane);
          acc0 = __builtin_amdgcn_wmma_f32_16x16x32_f16(
              false, a, false, b0, (short)0, acc0, false, false);
          v16h b1 = load_B(sWih, IN, 3 * HIDDEN + ct * 16, k0, lane);
          acc1 = __builtin_amdgcn_wmma_f32_16x16x32_f16(
              false, a, false, b1, (short)0, acc1, false, false);
        }
#pragma unroll
        for (int k0 = 0; k0 < PROJ; k0 += 32) {
          v16h a = load_A(sH, PROJ, m0, k0, lane);
          v16h b0 = load_B(sWhh, PROJ, 2 * HIDDEN + ct * 16, k0, lane);
          acc0 = __builtin_amdgcn_wmma_f32_16x16x32_f16(
              false, a, false, b0, (short)0, acc0, false, false);
          v16h b1 = load_B(sWhh, PROJ, 3 * HIDDEN + ct * 16, k0, lane);
          acc1 = __builtin_amdgcn_wmma_f32_16x16x32_f16(
              false, a, false, b1, (short)0, acc1, false, false);
        }
#pragma unroll
        for (int e = 0; e < 8; ++e) {
          float cv = fc[e] + ig[e] * tanh_(acc0[e] + biasr[p][2]);  // new c
          cst[p][e] = cv;
          float av = sigmoid_(acc1[e] + biasr[p][3]) * tanh_(cv);   // o*tanh(c)
          sA[(m0 + e + rowOff) * HIDDEN + col] = (_Float16)av;
        }
      }
    }
    __syncthreads();

    // ---- projection GEMM: h(128 x 64) = A(128x128) @ Whr^T ----
#pragma unroll
    for (int q = 0; q < 2; ++q) {
      const int tile = wave * 2 + q;
      const int mt = tile >> 2, nt = tile & 3;
      const int m0 = mt * 16, n0 = nt * 16;
      v8f acc = {};
#pragma unroll
      for (int k0 = 0; k0 < HIDDEN; k0 += 32) {
        v16h a = load_A(sA, HIDDEN, m0, k0, lane);
        v16h b = load_B(sWhr, HIDDEN, n0, k0, lane);
        acc = __builtin_amdgcn_wmma_f32_16x16x32_f16(
            false, a, false, b, (short)0, acc, false, false);
      }
      const int n = n0 + (lane & 15);
      const int rowOff = (lane >= 16) ? 8 : 0;
#pragma unroll
      for (int e = 0; e < 8; ++e) {
        const int m = m0 + e + rowOff;
        _Float16 hv = (_Float16)acc[e];
        sH[m * PROJ + n] = hv;                       // recurrent state for t+1
        const size_t oidx = ((size_t)t * BATCH + m) * (2 * PROJ) + dir * PROJ + n;
        if (outH) outH[oidx] = hv;                   // f16 feed to next layer
        if (outF) outF[oidx] = acc[e];               // final f32 output
      }
    }
    __syncthreads();
  }
}

// ---------------------------------------------------------------------------
static size_t lds_bytes(int in_dim) {
  size_t halves = (size_t)NGATES * in_dim       // Wih
                + (size_t)NGATES * PROJ         // Whh
                + (size_t)PROJ * HIDDEN         // Whr
                + (size_t)2 * BATCH * in_dim    // x_t ping-pong
                + (size_t)BATCH * PROJ          // h
                + (size_t)BATCH * HIDDEN;       // a = o*tanh(c)
  return halves * 2;                            // layer1: exactly 320 KB
}

extern "C" void kernel_launch(void* const* d_in, const int* in_sizes, int n_in,
                              void* d_out, int out_size, void* d_ws, size_t ws_size,
                              hipStream_t stream) {
  (void)in_sizes; (void)n_in; (void)out_size; (void)ws_size;
  const int*   x   = (const int*)d_in[0];
  const float* emb = (const float*)d_in[1];
  auto dp = [&](int b) {
    DirParams p;
    p.Wih = (const float*)d_in[b + 0];
    p.Whh = (const float*)d_in[b + 1];
    p.bih = (const float*)d_in[b + 2];
    p.bhh = (const float*)d_in[b + 3];
    p.Whr = (const float*)d_in[b + 4];
    return p;
  };
  DirParams l0f = dp(2), l0b = dp(7), l1f = dp(12), l1b = dp(17);

  _Float16* X0 = (_Float16*)d_ws;                       // (L,B,64)  f16  8 MB
  _Float16* X1 = X0 + (size_t)L_SEQ * BATCH * EMBED;    // (L,B,128) f16 16 MB
  float* out = (float*)d_out;

  embed_f16_kernel<<<(L_SEQ * BATCH * 8) / 256, 256, 0, stream>>>(x, emb, X0);
  lstm_scan_kernel<EMBED><<<2, 512, lds_bytes(EMBED), stream>>>(
      l0f, l0b, X0, X1, (float*)nullptr);
  lstm_scan_kernel<2 * PROJ><<<2, 512, lds_bytes(2 * PROJ), stream>>>(
      l1f, l1b, X1, (_Float16*)nullptr, out);
}
